// VSSBlock_DSM_4956392259737
// MI455X (gfx1250) — compile-verified, matrix operand order checked by reading, not run on hardware
//
#include <hip/hip_runtime.h>

// ---------------- problem constants ----------------
#define B_   8
#define C_   256
#define HW_  6400        // 80*80
#define NB_  51200       // B_*HW_
#define M_   1000
#define MP_  1024        // padded bank rows
#define KNN_ 20
#define NC_  8

typedef __attribute__((ext_vector_type(16))) __bf16 v16bf;
typedef __attribute__((ext_vector_type(8)))  __bf16 v8bf;
typedef __attribute__((ext_vector_type(8)))  float  v8f;
typedef __attribute__((ext_vector_type(4)))  int    v4i;

// ---------------- CDNA5 async-to-LDS staging (guarded) ----------------
#ifndef __has_builtin
#define __has_builtin(x) 0
#endif
#if __has_builtin(__builtin_amdgcn_global_load_async_to_lds_b128)
#define HAS_ASYNC_LDS 1
#else
#define HAS_ASYNC_LDS 0
#endif
#if __has_builtin(__builtin_amdgcn_s_wait_asynccnt)
#define HAS_WAIT_ASYNC 1
#else
#define HAS_WAIT_ASYNC 0
#endif

__device__ __forceinline__ void stage_chunk16(const __bf16* g, __bf16* l) {
#if HAS_ASYNC_LDS
  // emits GLOBAL_LOAD_ASYNC_TO_LDS_B128 (ASYNCcnt-tracked, ISA 10 op 98)
  __builtin_amdgcn_global_load_async_to_lds_b128(
      (__attribute__((address_space(1))) v4i*)g,
      (__attribute__((address_space(3))) v4i*)l,
      0, 0);
#else
  *(v8bf*)l = *(const v8bf*)g;
#endif
}

__device__ __forceinline__ void stage_fence() {
#if HAS_ASYNC_LDS
#if HAS_WAIT_ASYNC
  __builtin_amdgcn_s_wait_asynccnt(0);
#else
  asm volatile("s_wait_asynccnt 0x0" ::: "memory");
#endif
#endif
}

// build v16bf fragment from two contiguous 8-element halves
__device__ __forceinline__ v16bf mk16(v8bf lo, v8bf hi) {
  v16bf r;
#pragma unroll
  for (int e = 0; e < 8; e++) { r[e] = lo[e]; r[8 + e] = hi[e]; }
  return r;
}

// =====================================================================
// K0: transpose+convert x [B,C,HW] fp32 -> xT [NB,C] bf16 (pixel-major)
// =====================================================================
__global__ __launch_bounds__(256) void k_transpose(const float* __restrict__ x,
                                                   __bf16* __restrict__ xtb) {
  __shared__ float tile[32][33];
  int nt = blockIdx.x, ct = blockIdx.y, b = blockIdx.z;
  int n0 = nt * 32, c0 = ct * 32;
  int tx = threadIdx.x, ty = threadIdx.y;
  const float* xb = x + (size_t)b * C_ * HW_;
#pragma unroll
  for (int i = 0; i < 4; i++) {
    int c = c0 + ty + 8 * i;
    tile[ty + 8 * i][tx] = xb[(size_t)c * HW_ + n0 + tx];
  }
  __syncthreads();
#pragma unroll
  for (int i = 0; i < 4; i++) {
    int n = n0 + ty + 8 * i;
    xtb[((size_t)b * HW_ + n) * C_ + c0 + tx] = (__bf16)tile[tx][ty + 8 * i];
  }
}

// =====================================================================
// K0b: convert w_proj [256,256] fp32 -> bf16
// =====================================================================
__global__ __launch_bounds__(1024) void k_wconv(const float* __restrict__ w,
                                                __bf16* __restrict__ wb) {
  size_t i = (size_t)blockIdx.x * 1024 + threadIdx.x;
  wb[i] = (__bf16)w[i];
}

// =====================================================================
// K1: projection GEMM  y[n][o] = sum_c xT[n][c] * w[o][c]
//     16-pixel M-tile per block; each wave computes its 4 N-tiles
//     concurrently (4 accumulators) sharing one A fragment per K-step.
// =====================================================================
__global__ __launch_bounds__(128) void k_proj_gemm(const __bf16* __restrict__ xtb,
                                                   const __bf16* __restrict__ wb,
                                                   float* __restrict__ y) {
  __shared__ __bf16 sA[16][280];   // +24 pad: 16B-aligned rows, conflict-free
  int t = threadIdx.x;
  int lane = t & 31, wave = t >> 5;
  size_t n0 = (size_t)blockIdx.x * 16;

#pragma unroll
  for (int i = 0; i < 4; i++) {
    int id = t + 128 * i;
    int r  = id >> 5;
    int cc = (id & 31) * 8;
    stage_chunk16(xtb + (n0 + r) * C_ + cc, &sA[r][cc]);
  }
  stage_fence();
  __syncthreads();

  int ml   = lane & 15;              // M (A) / N (B) index within tile
  int koff = (lane < 16) ? 0 : 8;    // K sub-offset per lane half (ISA 7.12.2)
  int jt0  = wave * 4;

  v8f acc[4] = {};
#pragma unroll
  for (int kk = 0; kk < 8; kk++) {
    int kb = kk * 32;
    v16bf a = mk16(*(const v8bf*)&sA[ml][kb + koff],
                   *(const v8bf*)&sA[ml][kb + 16 + koff]);
#pragma unroll
    for (int u = 0; u < 4; u++) {
      const __bf16* brow = wb + (size_t)((jt0 + u) * 16 + ml) * C_ + kb + koff;
      v16bf bf = mk16(*(const v8bf*)brow, *(const v8bf*)(brow + 16));
      acc[u] = __builtin_amdgcn_wmma_f32_16x16x32_bf16(false, a, false, bf,
                                                       (short)0, acc[u], false, false);
    }
  }
#pragma unroll
  for (int u = 0; u < 4; u++) {
#pragma unroll
    for (int r = 0; r < 8; r++) {
      int m = r + ((lane >> 4) << 3);
      int j = (jt0 + u) * 16 + ml;
      y[(n0 + m) * C_ + j] = acc[u][r];
    }
  }
}

// =====================================================================
// K2: BN partial stats (sum, sumsq per channel) via coalesced tiles
// =====================================================================
__global__ __launch_bounds__(256) void k_bn_stats(const float* __restrict__ y,
                                                  float* __restrict__ sums) {
  int c = threadIdx.x;
  size_t n0 = (size_t)blockIdx.x * 256;
  float s = 0.f, s2 = 0.f;
  for (int p = 0; p < 256; p++) {
    float v = y[(n0 + p) * C_ + c];
    s += v; s2 += v * v;
  }
  atomicAdd(&sums[c], s);
  atomicAdd(&sums[C_ + c], s2);
}

// =====================================================================
// K3: finalize BN -> scale/shift
// =====================================================================
__global__ void k_bn_finalize(const float* __restrict__ sums,
                              const float* __restrict__ gamma,
                              const float* __restrict__ beta,
                              float* __restrict__ ss) {
  int c = threadIdx.x;
  float mean = sums[c] * (1.0f / (float)NB_);
  float var  = sums[C_ + c] * (1.0f / (float)NB_) - mean * mean;
  float sc   = gamma[c] * rsqrtf(var + 1e-5f);
  ss[c]      = sc;
  ss[C_ + c] = beta[c] - mean * sc;
}

// =====================================================================
// K4: normalize + ReLU -> feats fp32 (in place), bf16 copy, row norms a2
// =====================================================================
__global__ __launch_bounds__(256) void k_norm_relu(float* __restrict__ y,
                                                   const float* __restrict__ ss,
                                                   __bf16* __restrict__ fb,
                                                   float* __restrict__ a2) {
  __shared__ float red[256];
  int c = threadIdx.x;
  size_t n = blockIdx.x;
  float v = y[n * C_ + c];
  float f = fmaxf(v * ss[c] + ss[C_ + c], 0.0f);
  y[n * C_ + c]  = f;
  fb[n * C_ + c] = (__bf16)f;
  red[c] = f * f;
  __syncthreads();
  for (int ofs = 128; ofs > 0; ofs >>= 1) {
    if (c < ofs) red[c] += red[c + ofs];
    __syncthreads();
  }
  if (c == 0) a2[n] = red[0];
}

// =====================================================================
// K5: bank -> bf16 (padded to 1024 rows) + norms (pad rows get 1e12)
// =====================================================================
__global__ __launch_bounds__(256) void k_bank(const float* __restrict__ mb,
                                              __bf16* __restrict__ bb,
                                              float* __restrict__ b2) {
  __shared__ float red[256];
  int c = threadIdx.x;
  int j = blockIdx.x;
  float v = (j < M_) ? mb[(size_t)j * C_ + c] : 0.0f;
  bb[(size_t)j * C_ + c] = (__bf16)v;
  red[c] = v * v;
  __syncthreads();
  for (int ofs = 128; ofs > 0; ofs >>= 1) {
    if (c < ofs) red[c] += red[c + ofs];
    __syncthreads();
  }
  if (c == 0) b2[j] = (j < M_) ? red[0] : 1e12f;
}

// =====================================================================
// K6: FUSED dist-GEMM (bf16 WMMA, 4 concurrent N-tiles per wave)
//     + per-row top-20 + KDE density. 16 pixels x 1024 bank / block.
// =====================================================================
__global__ __launch_bounds__(128) void k_density(const __bf16* __restrict__ fb,
                                                 const __bf16* __restrict__ bb,
                                                 const float* __restrict__ a2,
                                                 const float* __restrict__ b2,
                                                 float* __restrict__ density) {
  __shared__ __bf16   sA[16][280];
  __shared__ _Float16 sD[16][MP_ + 8];
  __shared__ float    cand[16][8 * KNN_];
  __shared__ float    srk[16];
  __shared__ float    part[16][8];
  __shared__ float    sa2[16];

  int t = threadIdx.x, lane = t & 31, wave = t >> 5;
  size_t n0 = (size_t)blockIdx.x * 16;

#pragma unroll
  for (int i = 0; i < 4; i++) {
    int id = t + 128 * i;
    int r  = id >> 5;
    int cc = (id & 31) * 8;
    stage_chunk16(fb + (n0 + r) * C_ + cc, &sA[r][cc]);
  }
  if (t < 16) sa2[t] = a2[n0 + t];
  stage_fence();
  __syncthreads();

  int jl   = lane & 15;
  int koff = (lane < 16) ? 0 : 8;

  // --- phase 1: 16 N-tiles per wave, 4 concurrent ---
  for (int it = 0; it < 4; it++) {
    int jt0 = wave * 16 + it * 4;
    v8f acc[4] = {};
#pragma unroll
    for (int kk = 0; kk < 8; kk++) {
      int kb = kk * 32;
      if (kk < 7)
        __builtin_prefetch(bb + (size_t)(jt0 * 16 + jl) * C_ + kb + 32, 0, 3);
      v16bf a = mk16(*(const v8bf*)&sA[jl][kb + koff],
                     *(const v8bf*)&sA[jl][kb + 16 + koff]);
#pragma unroll
      for (int u = 0; u < 4; u++) {
        const __bf16* brow = bb + (size_t)((jt0 + u) * 16 + jl) * C_ + kb + koff;
        v16bf bf = mk16(*(const v8bf*)brow, *(const v8bf*)(brow + 16));
        acc[u] = __builtin_amdgcn_wmma_f32_16x16x32_bf16(false, a, false, bf,
                                                         (short)0, acc[u], false, false);
      }
    }
#pragma unroll
    for (int u = 0; u < 4; u++) {
      int j = (jt0 + u) * 16 + jl;
      float bj = b2[j];
#pragma unroll
      for (int r = 0; r < 8; r++) {
        int m = r + ((lane >> 4) << 3);
        float d2 = sa2[m] + bj - 2.0f * acc[u][r];
        sD[m][j] = (_Float16)sqrtf(fmaxf(d2, 0.0f));
      }
    }
  }
  __syncthreads();

  // --- phase 2: 8 threads per pixel row; local top-20 then merge ---
  int r = t >> 3, s = t & 7;
  float best[KNN_];
#pragma unroll
  for (int q = 0; q < KNN_; q++) best[q] = 3.4e38f;
  for (int i = 0; i < 128; i++) {
    float d = (float)sD[r][s + 8 * i];
    if (d < best[KNN_ - 1]) {
      int q = KNN_ - 1;
      while (q > 0 && best[q - 1] > d) { best[q] = best[q - 1]; q--; }
      best[q] = d;
    }
  }
#pragma unroll
  for (int q = 0; q < KNN_; q++) cand[r][s * KNN_ + q] = best[q];
  __syncthreads();

  if (s == 0) {
    float rk = 0.0f;
    for (int pass = 0; pass < KNN_; pass++) {
      float mn = 3.4e38f; int mi = 0;
      for (int q = 0; q < 8 * KNN_; q++) {
        float v = cand[r][q];
        if (v < mn) { mn = v; mi = q; }
      }
      cand[r][mi] = 3.4e38f;
      rk = mn;
    }
    srk[r] = fmaxf(rk, 1e-8f);   // ALPHA = 1
  }
  __syncthreads();

  float bw = srk[r];
  float inv_bw2 = 1.0f / (bw * bw);
  float dsum = 0.0f;
  for (int i = 0; i < 128; i++) {
    float d = (float)sD[r][s + 8 * i];
    dsum += __expf(-d * d * inv_bw2);
  }
  part[r][s] = dsum;
  __syncthreads();
  if (s == 0) {
    float tot = 0.f;
#pragma unroll
    for (int q = 0; q < 8; q++) tot += part[r][q];
    density[n0 + r] = tot;
  }
}

// =====================================================================
// K7: top-8 density peaks of LAST batch image -> centers, priors
// =====================================================================
__global__ __launch_bounds__(256) void k_peaks(const float* __restrict__ density,
                                               const float* __restrict__ feats,
                                               float* __restrict__ centers,
                                               float* __restrict__ cp) {
  __shared__ float cval[256 * NC_];
  __shared__ int   cidx[256 * NC_];
  __shared__ int   ti[NC_];
  int t = threadIdx.x;
  const float* drow = density + (size_t)(B_ - 1) * HW_;
  float bv[NC_]; int bi[NC_];
#pragma unroll
  for (int q = 0; q < NC_; q++) { bv[q] = -3.4e38f; bi[q] = 0; }
  for (int n = t; n < HW_; n += 256) {
    float v = drow[n];
    if (v > bv[NC_ - 1]) {
      int q = NC_ - 1;
      while (q > 0 && bv[q - 1] < v) { bv[q] = bv[q - 1]; bi[q] = bi[q - 1]; q--; }
      bv[q] = v; bi[q] = n;
    }
  }
#pragma unroll
  for (int q = 0; q < NC_; q++) { cval[t * NC_ + q] = bv[q]; cidx[t * NC_ + q] = bi[q]; }
  __syncthreads();
  if (t == 0) {
    float tv[NC_];
    for (int pass = 0; pass < NC_; pass++) {
      float mx = -3.4e38f; int mi = 0;
      for (int q = 0; q < 256 * NC_; q++)
        if (cval[q] > mx) { mx = cval[q]; mi = q; }
      tv[pass] = mx; ti[pass] = cidx[mi];
      cval[mi] = -3.4e38f;
    }
    float ssum = 0.f;
    for (int q = 0; q < NC_; q++) ssum += tv[q];
    for (int q = 0; q < NC_; q++) {
      cp[q]       = tv[q];
      cp[NC_ + q] = tv[q] / (ssum + 1e-8f);   // priors
    }
  }
  __syncthreads();
  for (int q = 0; q < NC_; q++)
    centers[q * C_ + t] = feats[((size_t)(B_ - 1) * HW_ + ti[q]) * C_ + t];
}

// =====================================================================
// K8: soft assignment to centers, density-prior weighted -> output
// =====================================================================
__global__ __launch_bounds__(256) void k_semantic(const float* __restrict__ feats,
                                                  const float* __restrict__ a2,
                                                  const float* __restrict__ centers,
                                                  const float* __restrict__ cp,
                                                  float* __restrict__ out) {
  __shared__ float sC[NC_][C_];
  __shared__ float c2[NC_];
  __shared__ float spri[NC_];
  int t = threadIdx.x;
#pragma unroll
  for (int q = 0; q < NC_; q++) sC[q][t] = centers[q * C_ + t];
  if (t < NC_) {
    float s = 0.f;
    for (int c = 0; c < C_; c++) { float v = centers[t * C_ + c]; s += v * v; }
    c2[t]   = s;
    spri[t] = cp[NC_ + t];
  }
  __syncthreads();

  size_t n = (size_t)blockIdx.x * 256 + t;
  const float* frow = feats + n * C_;
  float dot[NC_];
#pragma unroll
  for (int q = 0; q < NC_; q++) dot[q] = 0.f;
  for (int c = 0; c < C_; c += 4) {
    float4 f = *(const float4*)(frow + c);
#pragma unroll
    for (int q = 0; q < NC_; q++)
      dot[q] += f.x * sC[q][c] + f.y * sC[q][c + 1] + f.z * sC[q][c + 2] + f.w * sC[q][c + 3];
  }
  float an = a2[n];
  float logits[NC_];
  float mx = -3.4e38f;
#pragma unroll
  for (int q = 0; q < NC_; q++) {
    float d2 = fmaxf(an + c2[q] - 2.0f * dot[q], 0.0f);
    logits[q] = -sqrtf(d2) * 10.0f;          // 1/TEMPERATURE = 10
    mx = fmaxf(mx, logits[q]);
  }
  float se = 0.f, acc = 0.f;
#pragma unroll
  for (int q = 0; q < NC_; q++) {
    float e = __expf(logits[q] - mx);
    se  += e;
    acc += e * spri[q];
  }
  out[n] = acc / se;
}

// =====================================================================
// launch
// =====================================================================
extern "C" void kernel_launch(void* const* d_in, const int* in_sizes, int n_in,
                              void* d_out, int out_size, void* d_ws, size_t ws_size,
                              hipStream_t stream) {
  const float* x     = (const float*)d_in[0];
  const float* w     = (const float*)d_in[1];
  const float* gamma = (const float*)d_in[2];
  const float* beta  = (const float*)d_in[3];
  const float* mb    = (const float*)d_in[4];

  char* ws = (char*)d_ws;
  size_t off = 0;
  float*  y   = (float*) (ws + off); off += (size_t)NB_ * C_ * 4;   // y / feats fp32
  __bf16* fb  = (__bf16*)(ws + off); off += (size_t)NB_ * C_ * 2;   // xT bf16, then feats bf16
  __bf16* bb  = (__bf16*)(ws + off); off += (size_t)MP_ * C_ * 2;   // bank bf16 (padded)
  __bf16* wb  = (__bf16*)(ws + off); off += (size_t)C_  * C_ * 2;   // w bf16
  float*  b2  = (float*) (ws + off); off += (size_t)MP_ * 4;        // bank norms
  float*  a2  = (float*) (ws + off); off += (size_t)NB_ * 4;        // feat row norms
  float*  sums= (float*) (ws + off); off += 2 * C_ * 4;             // BN sum/sumsq
  float*  ss  = (float*) (ws + off); off += 2 * C_ * 4;             // BN scale/shift
  float*  dens= (float*) (ws + off); off += (size_t)NB_ * 4;        // density
  float*  ctr = (float*) (ws + off); off += NC_ * C_ * 4;           // centers
  float*  cp  = (float*) (ws + off); off += 2 * NC_ * 4;            // cdens + priors

  (void)hipMemsetAsync(sums, 0, 2 * C_ * 4, stream);

  k_transpose  <<<dim3(HW_ / 32, C_ / 32, B_), dim3(32, 8), 0, stream>>>(x, fb);
  k_wconv      <<<(C_ * C_) / 1024, 1024, 0, stream>>>(w, wb);
  k_proj_gemm  <<<NB_ / 16, 128, 0, stream>>>(fb, wb, y);
  k_bn_stats   <<<NB_ / 256, 256, 0, stream>>>(y, sums);
  k_bn_finalize<<<1, 256, 0, stream>>>(sums, gamma, beta, ss);
  k_norm_relu  <<<NB_, 256, 0, stream>>>(y, ss, fb, a2);
  k_bank       <<<MP_, 256, 0, stream>>>(mb, bb, b2);
  k_density    <<<NB_ / 16, 128, 0, stream>>>(fb, bb, a2, b2, dens);
  k_peaks      <<<1, 256, 0, stream>>>(dens, y, ctr, cp);
  k_semantic   <<<NB_ / 256, 256, 0, stream>>>(y, a2, ctr, cp, (float*)d_out);
}
